// TrinityGuidance_77335181132479
// MI455X (gfx1250) — matrix operand order repaired and checked by reading, not compiled, not run on hardware
//
#include <hip/hip_runtime.h>
#include <math.h>

// ---------------- problem constants (match reference) ----------------
#define GAMMA_C   10.0f
#define MGRID     64
#define THRESH_C  1.0f
#define SIGMA_C   1.5f
#define K_SOFT_C  2.0f
#define W_WL_C    1.0f
#define W_CONG_C  0.5f
#define BATCH     4
#define NVERT     20000
#define NPIN      400000
#define NNETS     50000
#define MAXPIN    32

// k_net tiling
#define NBLK      64      // blocks per batch for the net kernel
#define PANEL     128     // nets per LDS panel (== blockDim)
#define KPITCH    136     // K pitch in halves: 128 + 8 pad -> 16B-aligned runs,
                          // 272B row stride (68 banks) breaks power-of-2 conflicts

typedef __attribute__((ext_vector_type(16))) _Float16 v16h;
typedef __attribute__((ext_vector_type(8)))  float    v8f;
typedef __attribute__((ext_vector_type(4)))  float    v4f;

union HFrag { v16h v; v4f q[2]; _Float16 h[16]; };

// ---------------- kernel 0: zero accumulators ----------------
__global__ void tg_init_kernel(float* __restrict__ buf, int n) {
    int i = blockIdx.x * blockDim.x + threadIdx.x;
    if (i < n) buf[i] = 0.0f;
}

// ---------------- kernel 1: pin positions (gather + rotate) ----------------
// R0=I, R1=[[0,-1],[1,0]], R2=-I, R3=[[0,1],[-1,0]]
__global__ __launch_bounds__(256)
void tg_pinpos_kernel(const float* __restrict__ positions,     // [B][V][2]
                      const int*   __restrict__ pin_to_macro,  // [P]
                      const float* __restrict__ pin_offsets,   // [P][2]
                      const float* __restrict__ rot,           // [B][V][4]
                      float*       __restrict__ pin_pos)       // [B][P][2]
{
    long id = (long)blockIdx.x * blockDim.x + threadIdx.x;
    if (id >= (long)BATCH * NPIN) return;
    int b = (int)(id / NPIN);
    int p = (int)(id % NPIN);
    int m = pin_to_macro[p];
    float ox = pin_offsets[2 * p + 0];
    float oy = pin_offsets[2 * p + 1];
    const float* r = rot + ((long)b * NVERT + m) * 4;
    float r0 = r[0], r1 = r[1], r2 = r[2], r3 = r[3];
    float rx = (r0 - r2) * ox + (r3 - r1) * oy;
    float ry = (r1 - r3) * ox + (r0 - r2) * oy;
    const float* pp = positions + ((long)b * NVERT + m) * 2;
    pin_pos[id * 2 + 0] = pp[0] + rx;
    pin_pos[id * 2 + 1] = pp[1] + ry;
}

__device__ __forceinline__ float tg_sigmoid(float x) {
    return 1.0f / (1.0f + __expf(-x));
}

// ---------------- kernel 2: per-net LSE wirelength + RUDY GEMM via WMMA ----
// rudy[b,i,j] = sum_n (in_y[n,i]/size[n]) * in_x[n,j]   -> 64x64 = A^T(64xK) * B(Kx64)
// Operand panels staged in LDS in grid-major / K-minor layout so each WMMA
// fragment is two 16-byte ds_load_b128 runs:
//   A-frag (lane m, group lg): k(h) = lg*16 + h                 -> runs at
//        [col][kb + lg*16] and [col][kb + lg*16 + 8]
//   B-frag (lane n, group lg): k(h) = lg*16 + h                 -> runs at
//        [col][kb + lg*16] and [col][kb + lg*16 + 8]
__global__ __launch_bounds__(128)
void tg_net_kernel(const float* __restrict__ pin_pos,     // [B][P][2]
                   const int*   __restrict__ net_to_pin,  // [NN][32]
                   const float* __restrict__ net_weights, // [NN]
                   float*       __restrict__ rudy,        // [B][64][64]
                   float*       __restrict__ hpwl,        // [B]
                   int nets_per_block)
{
    __shared__ __align__(16) _Float16 lAyT[MGRID][KPITCH];  // in_y/size, [i][net]
    __shared__ __align__(16) _Float16 lAxT[MGRID][KPITCH];  // in_x,      [j][net]
    __shared__ float sred[PANEL];

    const int tid  = threadIdx.x;
    const int b    = blockIdx.x / NBLK;
    const int blk  = blockIdx.x % NBLK;
    const int base = blk * nets_per_block;
    const int wv   = tid >> 5;           // wave id 0..3 -> i-strip
    const int lane = tid & 31;
    const int lg   = lane >> 4;          // lane group (0: lanes 0-15, 1: 16-31)
    const int lm   = lane & 15;

    const int panels = (nets_per_block + PANEL - 1) / PANEL;

    v8f acc0 = {}, acc1 = {}, acc2 = {}, acc3 = {};
    float hp_acc = 0.0f;

    const float* batch_pins = pin_pos + (long)b * NPIN * 2;

    for (int pnl = 0; pnl < panels; ++pnl) {
        const int local = pnl * PANEL + tid;
        const int g = base + local;
        const bool valid = (local < nets_per_block) && (g < NNETS);

        // ---------- phase A: per-net scalar work ----------
        if (valid) {
            const int* row = net_to_pin + (long)g * MAXPIN;
            float maxx = -1e30f, minx = 1e30f, maxy = -1e30f, miny = 1e30f;
            // pass 1: bbox
            for (int j = 0; j < MAXPIN; ++j) {
                int pi = row[j];
                if (pi < 0) continue;
                float px = batch_pins[(long)pi * 2 + 0];
                float py = batch_pins[(long)pi * 2 + 1];
                maxx = fmaxf(maxx, px); minx = fminf(minx, px);
                maxy = fmaxf(maxy, py); miny = fminf(miny, py);
            }
            // pass 2: stabilized logsumexp sums (cache-hot re-gather)
            float sxp = 0.f, sxn = 0.f, syp = 0.f, syn = 0.f;
            for (int j = 0; j < MAXPIN; ++j) {
                int pi = row[j];
                if (pi < 0) continue;
                float px = batch_pins[(long)pi * 2 + 0];
                float py = batch_pins[(long)pi * 2 + 1];
                sxp += __expf(GAMMA_C * (px - maxx));
                sxn += __expf(-GAMMA_C * (px - minx));
                syp += __expf(GAMMA_C * (py - maxy));
                syn += __expf(-GAMMA_C * (py - miny));
            }
            float lse_max_x = maxx + __logf(sxp) * (1.0f / GAMMA_C);
            float lse_min_x = minx - __logf(sxn) * (1.0f / GAMMA_C);
            float lse_max_y = maxy + __logf(syp) * (1.0f / GAMMA_C);
            float lse_min_y = miny - __logf(syn) * (1.0f / GAMMA_C);
            float wl = (lse_max_x - lse_min_x) + (lse_max_y - lse_min_y);
            hp_acc += wl * net_weights[g];

            // soft bbox window on the 64-grid
            const float sc = 0.5f * (float)(MGRID - 1);
            float bxmin = (minx + 1.0f) * sc, bxmax = (maxx + 1.0f) * sc;
            float bymin = (miny + 1.0f) * sc, bymax = (maxy + 1.0f) * sc;
            float bsz = fmaxf((bxmax - bxmin + 1.0f) * (bymax - bymin + 1.0f), 1.0f);
            float inv_sz = 1.0f / bsz;
            for (int gi = 0; gi < MGRID; ++gi) {
                float fg = (float)gi;
                float fx = tg_sigmoid(K_SOFT_C * (fg - bxmin + 0.5f)) *
                           tg_sigmoid(K_SOFT_C * (bxmax - fg + 0.5f));
                float fy = tg_sigmoid(K_SOFT_C * (fg - bymin + 0.5f)) *
                           tg_sigmoid(K_SOFT_C * (bymax - fg + 0.5f));
                lAxT[gi][tid] = (_Float16)fx;                 // coalesced across lanes
                lAyT[gi][tid] = (_Float16)(fy * inv_sz);
            }
        } else {
            for (int gi = 0; gi < MGRID; ++gi) {
                lAxT[gi][tid] = (_Float16)0.0f;
                lAyT[gi][tid] = (_Float16)0.0f;
            }
        }
        __syncthreads();

        // ---------- phase B: WMMA over the 128-net panel (K = 4 x 32) ----------
        {
            const int acol = wv * 16 + lm;   // A column (output row i)
            for (int ks = 0; ks < 4; ++ks) {
                const int kb = ks * 32;
                HFrag a;
                a.q[0] = *(const v4f*)&lAyT[acol][kb + lg * 16];
                a.q[1] = *(const v4f*)&lAyT[acol][kb + lg * 16 + 8];

                HFrag b0, b1, b2, b3;
                const int kB = kb + lg * 16;
                b0.q[0] = *(const v4f*)&lAxT[ 0 + lm][kB];
                b0.q[1] = *(const v4f*)&lAxT[ 0 + lm][kB + 8];
                b1.q[0] = *(const v4f*)&lAxT[16 + lm][kB];
                b1.q[1] = *(const v4f*)&lAxT[16 + lm][kB + 8];
                b2.q[0] = *(const v4f*)&lAxT[32 + lm][kB];
                b2.q[1] = *(const v4f*)&lAxT[32 + lm][kB + 8];
                b3.q[0] = *(const v4f*)&lAxT[48 + lm][kB];
                b3.q[1] = *(const v4f*)&lAxT[48 + lm][kB + 8];

                acc0 = __builtin_amdgcn_wmma_f32_16x16x32_f16(false, a.v, false, b0.v, (short)0, acc0, false, false);
                acc1 = __builtin_amdgcn_wmma_f32_16x16x32_f16(false, a.v, false, b1.v, (short)0, acc1, false, false);
                acc2 = __builtin_amdgcn_wmma_f32_16x16x32_f16(false, a.v, false, b2.v, (short)0, acc2, false, false);
                acc3 = __builtin_amdgcn_wmma_f32_16x16x32_f16(false, a.v, false, b3.v, (short)0, acc3, false, false);
            }
        }
        __syncthreads();
    }

    // ---------- flush C fragments: 16x16 f32 layout (lane->N, vgpr d -> M) ----------
    float* rb = rudy + (long)b * MGRID * MGRID;
    #pragma unroll
    for (int d = 0; d < 8; ++d) {
        int i = wv * 16 + d + 8 * lg;
        atomicAdd(&rb[i * MGRID +  0 + lm], acc0[d]);
        atomicAdd(&rb[i * MGRID + 16 + lm], acc1[d]);
        atomicAdd(&rb[i * MGRID + 32 + lm], acc2[d]);
        atomicAdd(&rb[i * MGRID + 48 + lm], acc3[d]);
    }

    // ---------- block-reduce weighted wirelength ----------
    sred[tid] = hp_acc;
    __syncthreads();
    for (int s = PANEL / 2; s > 0; s >>= 1) {
        if (tid < s) sred[tid] += sred[tid + s];
        __syncthreads();
    }
    if (tid == 0) atomicAdd(&hpwl[b], sred[0]);
}

// ---------------- kernel 3: 7x7 Gaussian smooth + overflow penalty + combine
__global__ __launch_bounds__(256)
void tg_conv_kernel(const float* __restrict__ rudy,  // [B][64][64]
                    const float* __restrict__ hpwl,  // [B]
                    float*       __restrict__ out)   // [B]
{
    __shared__ float tile[MGRID][MGRID + 1];
    __shared__ float red[256];
    const int b = blockIdx.x;
    const int tid = threadIdx.x;

    for (int idx = tid; idx < MGRID * MGRID; idx += 256)
        tile[idx >> 6][idx & 63] = rudy[(long)b * MGRID * MGRID + idx];
    __syncthreads();

    // separable gaussian weights, ksize = 7 (sigma 1.5), normalized
    float g[7]; float gs = 0.0f;
    #pragma unroll
    for (int i = 0; i < 7; ++i) {
        float x = (float)(i - 3);
        g[i] = __expf(-x * x / (2.0f * SIGMA_C * SIGMA_C));
        gs += g[i];
    }
    const float inv_norm = 1.0f / (gs * gs);

    float pen = 0.0f;
    for (int idx = tid; idx < MGRID * MGRID; idx += 256) {
        int i = idx >> 6, j = idx & 63;
        float s = 0.0f;
        #pragma unroll
        for (int dy = -3; dy <= 3; ++dy) {
            int ii = i + dy;
            if (ii < 0 || ii >= MGRID) continue;  // zero padding
            float gy = g[dy + 3];
            #pragma unroll
            for (int dx = -3; dx <= 3; ++dx) {
                int jj = j + dx;
                if (jj < 0 || jj >= MGRID) continue;
                s += tile[ii][jj] * gy * g[dx + 3];
            }
        }
        s *= inv_norm;
        float ov = fmaxf(s - THRESH_C, 0.0f);
        pen += ov * ov;
    }

    red[tid] = pen;
    __syncthreads();
    for (int s2 = 128; s2 > 0; s2 >>= 1) {
        if (tid < s2) red[tid] += red[tid + s2];
        __syncthreads();
    }
    if (tid == 0) out[b] = W_WL_C * hpwl[b] + W_CONG_C * red[0];
}

// ---------------- host-side orchestration ----------------
extern "C" void kernel_launch(void* const* d_in, const int* in_sizes, int n_in,
                              void* d_out, int out_size, void* d_ws, size_t ws_size,
                              hipStream_t stream) {
    const float* positions    = (const float*)d_in[0];  // [B][V][2]
    const int*   net_to_pin   = (const int*)  d_in[1];  // [NN][32]
    const int*   pin_to_macro = (const int*)  d_in[2];  // [P]
    const float* pin_offsets  = (const float*)d_in[3];  // [P][2]
    const float* rotation_oh  = (const float*)d_in[4];  // [B][V][4]
    const float* net_weights  = (const float*)d_in[5];  // [NN]
    float* out = (float*)d_out;                          // [B]

    // workspace layout
    float* pin_pos = (float*)d_ws;                               // B*P*2 floats
    float* rudy    = pin_pos + (size_t)BATCH * NPIN * 2;         // B*64*64 floats
    float* hpwl    = rudy + (size_t)BATCH * MGRID * MGRID;       // B floats
    (void)in_sizes; (void)n_in; (void)out_size; (void)ws_size;

    // zero accumulators (rudy + hpwl are contiguous)
    const int n_init = BATCH * MGRID * MGRID + BATCH;
    tg_init_kernel<<<(n_init + 255) / 256, 256, 0, stream>>>(rudy, n_init);

    // pin positions
    const long n_pp = (long)BATCH * NPIN;
    tg_pinpos_kernel<<<(int)((n_pp + 255) / 256), 256, 0, stream>>>(
        positions, pin_to_macro, pin_offsets, rotation_oh, pin_pos);

    // per-net wirelength + WMMA rudy accumulation
    const int nets_per_block = (NNETS + NBLK - 1) / NBLK;
    tg_net_kernel<<<BATCH * NBLK, PANEL, 0, stream>>>(
        pin_pos, net_to_pin, net_weights, rudy, hpwl, nets_per_block);

    // smooth + penalty + combine
    tg_conv_kernel<<<BATCH, 256, 0, stream>>>(rudy, hpwl, out);
}
